// RelativeMultiHeadSelfAttention_56513179680960
// MI455X (gfx1250) — matrix operand order repaired and checked by reading, not compiled, not run on hardware
//
#include <hip/hip_runtime.h>
#include <hip/hip_bf16.h>

typedef __attribute__((ext_vector_type(16))) _Float16     v16h;
typedef __attribute__((ext_vector_type(8)))  _Float16     v8h;
typedef __attribute__((ext_vector_type(8)))  float        v8f;
typedef __attribute__((ext_vector_type(4)))  unsigned int u32x4;
typedef __attribute__((ext_vector_type(8)))  int          i32x8;
typedef __attribute__((ext_vector_type(4)))  int          i32x4;

#define NB    8
#define NTOK  784
#define NHEAD 12
#define HDIM  32
#define CIN   384
#define HID   384
#define BN    (NB*NTOK)               // 6272 rows
#define WSZ   (CIN*HID)               // 147456 elements per weight matrix
#define SCALE 0.05103103630798288f    // 384^-0.5 (reference scales by hidden^-0.5)

__device__ __forceinline__ v16h frag16(const _Float16* p0, const _Float16* p1) {
  const v8h a0 = *(const v8h*)p0;
  const v8h a1 = *(const v8h*)p1;
  return __builtin_shufflevector(a0, a1, 0,1,2,3,4,5,6,7,8,9,10,11,12,13,14,15);
}

// ---------------------------------------------------------------------------
// Kernel 0: one-time weight prep: WT[mat][n][k] = (f16) W_mat[k][n].
// Makes every WMMA B-fragment a contiguous 32B load.
// ---------------------------------------------------------------------------
__global__ __launch_bounds__(256) void wprep_kernel(
    const float* __restrict__ Wq, const float* __restrict__ Wk,
    const float* __restrict__ Wv, const float* __restrict__ Wo,
    _Float16* __restrict__ wt)
{
  const int idx = blockIdx.x * 256 + threadIdx.x;
  if (idx >= 4 * WSZ) return;
  const int m = idx / WSZ;
  const int r = idx % WSZ;
  const int n = r / CIN, k = r % CIN;
  const float* W = (m == 0) ? Wq : ((m == 1) ? Wk : ((m == 2) ? Wv : Wo));
  wt[idx] = (_Float16)W[(size_t)k * HID + n];
}

// ---------------------------------------------------------------------------
// TDM helper: 1-D global->LDS DMA, n_u64 8-byte elements (contiguous).
// D# packing per CDNA5 ISA (groups 0/1; rest zero).
// This toolchain exposes the 6-arg builtin form:
//   (u32x4 g0, i32x8 g1, i32x4 g2, i32x4 g3, i32x8 g4, i32 cpol)
// ---------------------------------------------------------------------------
#if __has_builtin(__builtin_amdgcn_tensor_load_to_lds) && __has_builtin(__builtin_amdgcn_s_wait_tensorcnt)
#define HAVE_TDM 1
__device__ __forceinline__ void tdm_load_1d(const void* gsrc, void* ldsdst, unsigned n_u64) {
  const unsigned long long ga  = (unsigned long long)(size_t)gsrc;
  const unsigned           lds = (unsigned)(size_t)ldsdst;     // LDS aperture: offset in addr[31:0]
  u32x4 g0;
  g0[0] = 1u;                                                  // count=1, user mode
  g0[1] = lds;                                                 // lds_addr  (bits 63:32)
  g0[2] = (unsigned)(ga & 0xFFFFFFFFull);                      // global_addr[31:0]
  g0[3] = (unsigned)((ga >> 32) & 0x1FFFFFFull) | (2u << 30);  // global_addr[56:32] | type=2
  // group1 bit layout (bit index within 256b):
  //  data_size@16(=3: 8B)  tensor_dim0@48(32b)  tensor_dim1@80(32b)=1
  //  tile_dim0@112(16b)    tile_dim1@128=0      tile_dim2@144=0
  //  dim0_stride@160(48b)  dim1_stride@208(48b)
  const unsigned long long nd = n_u64;
  const unsigned long long q0 = (3ull << 16) | ((nd & 0xFFFFull) << 48);
  const unsigned long long q1 = (nd >> 16) | (1ull << 16) | ((nd & 0xFFFFull) << 48);
  const unsigned long long q2 = (nd & 0xFFFFFFFFull) << 32;    // dim0_stride[31:0]
  const unsigned long long q3 = nd << 16;                      // dim1_stride (don't-care, benign)
  i32x8 g1;
  g1[0] = (int)q0; g1[1] = (int)(q0 >> 32);
  g1[2] = (int)q1; g1[3] = (int)(q1 >> 32);
  g1[4] = (int)q2; g1[5] = (int)(q2 >> 32);
  g1[6] = (int)q3; g1[7] = (int)(q3 >> 32);
  const i32x4 z4 = {0, 0, 0, 0};
  const i32x8 z8 = {0, 0, 0, 0, 0, 0, 0, 0};
  __builtin_amdgcn_tensor_load_to_lds(g0, g1, z4, z4, z8, 0);
}
#else
#define HAVE_TDM 0
#endif

// ---------------------------------------------------------------------------
// Kernel 1: fused QKV projection.  X(6272x384,f32) x WT(q,k,v) -> per-head f16
// q/k/v[(b*12+h)*784 + n][32].
// ---------------------------------------------------------------------------
__global__ __launch_bounds__(256) void qkv_kernel(
    const float* __restrict__ x, const _Float16* __restrict__ wt,
    const float* __restrict__ bq, const float* __restrict__ bk, const float* __restrict__ bv,
    _Float16* __restrict__ qws, _Float16* __restrict__ kws, _Float16* __restrict__ vws)
{
  __shared__ __align__(32) _Float16 Alds[16 * CIN];   // 12 KB
  const int mt  = blockIdx.x;                         // 16-row tile
  const int tid = threadIdx.x;

  for (int i = tid; i < 16 * CIN; i += 256)
    Alds[i] = (_Float16)x[(size_t)(mt * 16 + (i / CIN)) * CIN + (i % CIN)];
  __syncthreads();

  const int  wave = tid >> 5, lane = tid & 31;
  const int  lo   = lane & 15;
  const bool hih  = lane >= 16;

  for (int t = wave; t < 72; t += 8) {                // 3 matrices x 24 col tiles
    const int mat  = t / 24;
    const int col0 = (t % 24) * 16;
    const _Float16* Wm = wt + (size_t)mat * WSZ;      // WT[n][k]
    const float*    bm = (mat == 0) ? bq : ((mat == 1) ? bk : bv);

    v8f acc = {};
    for (int kt = 0; kt < 12; ++kt) {
      const int k0 = kt * 32;
      const _Float16* ap = &Alds[lo * CIN + k0 + (hih ? 16 : 0)];
      const v16h a = frag16(ap, ap + 8);
      const _Float16* wp = &Wm[(size_t)(col0 + lo) * CIN + k0 + (hih ? 16 : 0)];
      __builtin_prefetch(wp + 32, 0, 0);
      const v16h bf = *(const v16h*)wp;               // contiguous 32B
      acc = __builtin_amdgcn_wmma_f32_16x16x32_f16(false, a, false, bf,
                                                   (short)0, acc, false, false);
    }
    const float bias = bm[col0 + lo];
    const int   hcol = (col0 + lo) >> 5;
    const int   d    = (col0 + lo) & 31;
    _Float16* dst = (mat == 0) ? qws : ((mat == 1) ? kws : vws);
    #pragma unroll
    for (int r = 0; r < 8; ++r) {
      const int row = mt * 16 + r + (hih ? 8 : 0);
      const int bb = row / NTOK, nn = row % NTOK;
      dst[(((size_t)bb * NHEAD + hcol) * NTOK + nn) * HDIM + d] = (_Float16)(acc[r] + bias);
    }
  }
}

// ---------------------------------------------------------------------------
// Kernel 2: flash attention per (b, h, 112-row block).  7 waves x 16 rows.
// K staged via TDM DMA (784x32), V staged transposed (32x800 padded), online
// softmax with relative-position bias gather; scores never hit memory.
// ---------------------------------------------------------------------------
__global__ __launch_bounds__(224) void attn_kernel(
    const _Float16* __restrict__ qws, const _Float16* __restrict__ kws,
    const _Float16* __restrict__ vws,
    const float* __restrict__ table, const int* __restrict__ rel_index,
    _Float16* __restrict__ aws)
{
  __shared__ __align__(32) _Float16 Klds[NTOK * HDIM];   // 50176 B
  __shared__ __align__(32) _Float16 Vt[HDIM * 800];      // 51200 B (padded keys)
  __shared__ __align__(32) _Float16 Pld[7][512];         // per-wave P staging
  const int bh  = blockIdx.y;
  const int b   = bh / NHEAD, h = bh % NHEAD;
  const int tid = threadIdx.x;
  const size_t base = (size_t)bh * NTOK * HDIM;

  const int wave = tid >> 5, lane = tid & 31;
#if HAVE_TDM
  if (wave == 0) {                                       // one DMA per workgroup
    tdm_load_1d(kws + base, Klds, (NTOK * HDIM * 2) / 8);
    __builtin_amdgcn_s_wait_tensorcnt(0);
  }
#else
  for (int i = tid; i < NTOK * HDIM; i += 224) Klds[i] = kws[base + i];
#endif
  for (int i = tid; i < NTOK * HDIM; i += 224) {
    const int n = i >> 5, d = i & 31;
    Vt[d * 800 + n] = vws[base + i];
  }
  for (int i = tid; i < HDIM * 16; i += 224)             // zero key pad 784..799
    Vt[(i >> 4) * 800 + NTOK + (i & 15)] = (_Float16)0.f;
  __syncthreads();

  const int  lo  = lane & 15;
  const bool hih = lane >= 16;
  const int  q0  = blockIdx.x * 112 + wave * 16;

  // Q A-fragment straight from global (f16, two contiguous 16B chunks)
  const _Float16* qp = &qws[base + (size_t)(q0 + lo) * HDIM + (hih ? 16 : 0)];
  const v16h qa = frag16(qp, qp + 8);

  float mrow[8], lrow[8];
  v8f o0 = {}, o1 = {};
  #pragma unroll
  for (int r = 0; r < 8; ++r) { mrow[r] = -3.0e38f; lrow[r] = 0.f; }
  _Float16* Pw = Pld[wave];

  for (int jp = 0; jp < 25; ++jp) {          // pairs of 16-key tiles (49 tiles + pad)
    #pragma unroll
    for (int s = 0; s < 2; ++s) {
      const int j = jp * 2 + s;
      if (j < 49) {
        const int key = j * 16 + lo;
        const v16h kb = *(const v16h*)&Klds[key * HDIM + (hih ? 16 : 0)];
        v8f z = {};
        v8f c = __builtin_amdgcn_wmma_f32_16x16x32_f16(false, qa, false, kb,
                                                       (short)0, z, false, false);
        #pragma unroll
        for (int r = 0; r < 8; ++r) {
          const int qrow = q0 + r + (hih ? 8 : 0);
          int ridx = rel_index[qrow * NTOK + key];
          ridx = ridx < 0 ? 0 : (ridx > 3024 ? 3024 : ridx);   // jnp.take mode='clip'
          float sc = c[r] * SCALE + table[ridx * NHEAD + h];
          float tmax = sc;                                      // row-max over 16 lanes
          tmax = fmaxf(tmax, __shfl_xor(tmax, 1, 32));
          tmax = fmaxf(tmax, __shfl_xor(tmax, 2, 32));
          tmax = fmaxf(tmax, __shfl_xor(tmax, 4, 32));
          tmax = fmaxf(tmax, __shfl_xor(tmax, 8, 32));
          const float mnew  = fmaxf(mrow[r], tmax);
          const float alpha = __expf(mrow[r] - mnew);
          const float p     = __expf(sc - mnew);
          float psum = p;
          psum += __shfl_xor(psum, 1, 32);
          psum += __shfl_xor(psum, 2, 32);
          psum += __shfl_xor(psum, 4, 32);
          psum += __shfl_xor(psum, 8, 32);
          lrow[r] = lrow[r] * alpha + psum;
          mrow[r] = mnew;
          o0[r] *= alpha; o1[r] *= alpha;
          Pw[(r + (hih ? 8 : 0)) * 32 + s * 16 + lo] = (_Float16)p;
        }
      } else {
        #pragma unroll
        for (int r = 0; r < 8; ++r)
          Pw[(r + (hih ? 8 : 0)) * 32 + 16 + lo] = (_Float16)0.f;
      }
    }
    // P (16x32) x V (32x32): two WMMAs for dims 0..15 / 16..31
    const _Float16* pp = &Pw[lo * 32 + (hih ? 16 : 0)];
    const v16h pa = frag16(pp, pp + 8);
    const int kk = jp * 32 + (hih ? 16 : 0);
    const v16h vb0 = *(const v16h*)&Vt[lo * 800 + kk];
    const v16h vb1 = *(const v16h*)&Vt[(16 + lo) * 800 + kk];
    o0 = __builtin_amdgcn_wmma_f32_16x16x32_f16(false, pa, false, vb0, (short)0, o0, false, false);
    o1 = __builtin_amdgcn_wmma_f32_16x16x32_f16(false, pa, false, vb1, (short)0, o1, false, false);
  }

  #pragma unroll
  for (int r = 0; r < 8; ++r) {
    const int qrow = q0 + r + (hih ? 8 : 0);
    const float inv = 1.0f / lrow[r];
    const size_t rowbase = ((size_t)(b * NTOK + qrow)) * HID + h * HDIM;
    aws[rowbase + lo]      = (_Float16)(o0[r] * inv);
    aws[rowbase + 16 + lo] = (_Float16)(o1[r] * inv);
  }
}

// ---------------------------------------------------------------------------
// Kernel 3: output projection.  attn(6272x384,f16) x WoT + bo -> f32 out.
// ---------------------------------------------------------------------------
__global__ __launch_bounds__(256) void oproj_kernel(
    const _Float16* __restrict__ aws, const _Float16* __restrict__ wtO,
    const float* __restrict__ bo, float* __restrict__ out)
{
  __shared__ __align__(32) _Float16 Alds[16 * HID];
  const int mt  = blockIdx.x;
  const int tid = threadIdx.x;
  for (int i = tid; i < 16 * HID; i += 256)
    Alds[i] = aws[(size_t)mt * 16 * HID + i];
  __syncthreads();

  const int  wave = tid >> 5, lane = tid & 31;
  const int  lo   = lane & 15;
  const bool hih  = lane >= 16;

  for (int t = wave; t < 24; t += 8) {
    const int col0 = t * 16;
    v8f acc = {};
    for (int kt = 0; kt < 12; ++kt) {
      const int k0 = kt * 32;
      const _Float16* ap = &Alds[lo * HID + k0 + (hih ? 16 : 0)];
      const v16h a = frag16(ap, ap + 8);
      const _Float16* wp = &wtO[(size_t)(col0 + lo) * HID + k0 + (hih ? 16 : 0)];
      const v16h bf = *(const v16h*)wp;
      acc = __builtin_amdgcn_wmma_f32_16x16x32_f16(false, a, false, bf,
                                                   (short)0, acc, false, false);
    }
    const float bias = bo[col0 + lo];
    #pragma unroll
    for (int r = 0; r < 8; ++r) {
      const int row = mt * 16 + r + (hih ? 8 : 0);
      out[(size_t)row * HID + col0 + lo] = acc[r] + bias;
    }
  }
}

// ---------------------------------------------------------------------------
extern "C" void kernel_launch(void* const* d_in, const int* in_sizes, int n_in,
                              void* d_out, int out_size, void* d_ws, size_t ws_size,
                              hipStream_t stream) {
  const float* x     = (const float*)d_in[0];
  const float* Wq    = (const float*)d_in[1];
  const float* bq    = (const float*)d_in[2];
  const float* Wk    = (const float*)d_in[3];
  const float* bk    = (const float*)d_in[4];
  const float* Wv    = (const float*)d_in[5];
  const float* bv    = (const float*)d_in[6];
  const float* Wo    = (const float*)d_in[7];
  const float* bo    = (const float*)d_in[8];
  const float* table = (const float*)d_in[9];
  const int*   rel   = (const int*)d_in[10];

  const size_t perMat = (size_t)NB * NHEAD * NTOK * HDIM;   // 2,408,448 halfs
  _Float16* qws = (_Float16*)d_ws;
  _Float16* kws = qws + perMat;
  _Float16* vws = kws + perMat;
  _Float16* aws = vws + perMat;                             // [b*784+n][384]
  _Float16* wt  = aws + perMat;                             // 4 x 147456 halfs (q,k,v,o)

  wprep_kernel<<<(4 * WSZ + 255) / 256, 256, 0, stream>>>(Wq, Wk, Wv, Wo, wt);
  qkv_kernel<<<BN / 16, 256, 0, stream>>>(x, wt, bq, bk, bv, qws, kws, vws);
  attn_kernel<<<dim3(7, NB * NHEAD), 224, 0, stream>>>(qws, kws, vws, table, rel, aws);
  oproj_kernel<<<BN / 16, 256, 0, stream>>>(aws, wt + 3 * (size_t)WSZ, bo, (float*)d_out);
}
